// ScaledDotProduct_19164144075426
// MI455X (gfx1250) — matrix operand order repaired and checked by reading, not compiled, not run on hardware
//
#include <hip/hip_runtime.h>

// MI455X / gfx1250 flash-attention forward, block-cooperative K/V staging via
// GLOBAL_LOAD_ASYNC_TO_LDS_B128 (ASYNCcnt), double-buffered, WMMA f16 math.
//
// Fragment layouts (wave32):
//   A 16x32 f16 : lane row M = lane%16 ; VGPR v holds K = 16*(lane/16) + 2v, 2v+1
//   B 32x16 f16 : lane col N = lane%16 ; VGPR v holds K = 16*(lane/16) + 2v, 2v+1
//   C 16x16 f32 : lane col N = lane%16 ; VGPR v holds row M = v + 8*(lane/16)

typedef _Float16 half_t;
typedef __attribute__((ext_vector_type(16))) _Float16 v16h;
typedef __attribute__((ext_vector_type(2)))  _Float16 h2;
typedef __attribute__((ext_vector_type(8)))  float    v8f;
typedef __attribute__((ext_vector_type(4)))  int      v4i;

typedef __attribute__((address_space(1))) v4i gv4i;  // global b128 segment
typedef __attribute__((address_space(3))) v4i lv4i;  // LDS    b128 segment

#define BB 4
#define SS 2048
#define HH 16
#define DD 64
#define RS (HH * DD)        // float stride between sequence positions
#define WAVES 8             // waves (independent 16-query tiles) per block
#define CHUNK 32            // keys staged + processed per iteration
#define NCHUNK (SS / CHUNK)
#define LROW 68             // padded LDS row stride in floats (272B, 16B-aligned,
                            // bank = (4*key + dim) % 64 -> conflict-free gathers)

__device__ __forceinline__ int kpair_base(int v, int hlf) {
  return 16 * hlf + 2 * v;
}

// All 256 threads cooperatively issue the async copy of one 32-key chunk of K
// and V (8KB each) into LDS: 4 x b128 per thread, tracked with ASYNCcnt.
__device__ __forceinline__ void issue_chunk_copy(const float* kp, const float* vp,
                                                 int kb0, float* kdst, float* vdst,
                                                 int tid) {
  #pragma unroll
  for (int j = 0; j < 2; ++j) {
    const int s   = tid * 2 + j;      // 0..511 : 16B segment id
    const int row = s >> 4;           // key within chunk
    const int c4  = (s & 15) * 4;     // float offset within row
    const float* gk = kp + (size_t)(kb0 + row) * RS + c4;
    const float* gv = vp + (size_t)(kb0 + row) * RS + c4;
    __builtin_amdgcn_global_load_async_to_lds_b128(
        (gv4i*)gk, (lv4i*)(kdst + row * LROW + c4), 0, 0);
    __builtin_amdgcn_global_load_async_to_lds_b128(
        (gv4i*)gv, (lv4i*)(vdst + row * LROW + c4), 0, 0);
  }
}

__global__ __launch_bounds__(256) void attn_fwd_19164144075426(
    const float* __restrict__ Q, const float* __restrict__ K,
    const float* __restrict__ V, const unsigned char* __restrict__ M,
    float* __restrict__ O) {
  __shared__ float    kbuf[2][CHUNK * LROW];   // staged K chunk (f32)
  __shared__ float    vbuf[2][CHUNK * LROW];   // staged V chunk (f32)
  __shared__ _Float16 p_buf[WAVES][16 * 32];   // P^T staging, [q][key] f16
  __shared__ float    s_buf[WAVES][16];        // per-row scale / l broadcast

  const int tid  = threadIdx.x;
  const int wave = tid >> 5;
  const int lane = tid & 31;
  const int ln   = lane & 15;
  const int hlf  = lane >> 4;

  const int tile  = blockIdx.x * WAVES + wave;   // 8 consecutive q-tiles, same (b,h)
  const int qt    = tile & (SS / 16 - 1);
  const int bh    = tile / (SS / 16);
  const int h     = bh & (HH - 1);
  const int b     = bh / HH;
  const int qbase = qt * 16;

  const size_t bh_off = (size_t)b * SS * RS + (size_t)h * DD;  // + s*RS + d
  const float* qp = Q + bh_off;
  const float* kp = K + bh_off;
  const float* vp = V + bh_off;
  const unsigned char* mp = M + (size_t)(b * HH + h) * SS * SS;

  // ---- Q tile as two A fragments (d 0..31 / 32..63), pre-scaled by 0.125 ----
  v16h aq0, aq1;
  {
    const float* qrow = qp + (size_t)(qbase + ln) * RS;
    #pragma unroll
    for (int v = 0; v < 8; ++v) {
      const int d0 = kpair_base(v, hlf);
      aq0[2 * v]     = (half_t)(qrow[d0] * 0.125f);
      aq0[2 * v + 1] = (half_t)(qrow[d0 + 1] * 0.125f);
      aq1[2 * v]     = (half_t)(qrow[32 + d0] * 0.125f);
      aq1[2 * v + 1] = (half_t)(qrow[32 + d0 + 1] * 0.125f);
    }
  }

  // O^T accumulators: 4 chunks of 16 dims x 16 queries (rows=d, cols=q)
  v8f oacc[4];
  #pragma unroll
  for (int cc = 0; cc < 4; ++cc)
    #pragma unroll
    for (int i = 0; i < 8; ++i) oacc[cc][i] = 0.0f;

  float m_r[8], l_r[8];  // row stats, row = v + 8*hlf
  #pragma unroll
  for (int v = 0; v < 8; ++v) { m_r[v] = -1e30f; l_r[v] = 0.0f; }

  // prologue: start copy of chunk 0 into buffer 0
  issue_chunk_copy(kp, vp, 0, kbuf[0], vbuf[0], tid);

  int cur = 0;
  for (int ch = 0; ch < NCHUNK; ++ch) {
    const int kb0 = ch * CHUNK;

    // overlap: start chunk ch+1 into the other buffer (safe: last read of that
    // buffer was chunk ch-1, protected by the trailing barrier of that iter).
    if (ch + 1 < NCHUNK) {
      issue_chunk_copy(kp, vp, kb0 + CHUNK, kbuf[cur ^ 1], vbuf[cur ^ 1], tid);
      __builtin_amdgcn_s_wait_asynccnt(4);  // 4 older ops = chunk ch, in-order
    } else {
      __builtin_amdgcn_s_wait_asynccnt(0);
    }
    __syncthreads();  // everyone's slices of chunk ch have landed

    // token prefetch of next chunk's mask rows (dominant HBM stream)
    if (ch + 1 < NCHUNK)
      __builtin_prefetch(mp + (size_t)(qbase + ln) * SS + kb0 + CHUNK, 0, 1);

    // ---- scores: two 16x16 tiles (key subtiles t=0,1), fp32 accum ----
    v8f st[2];
    #pragma unroll
    for (int t = 0; t < 2; ++t) {
      #pragma unroll
      for (int i = 0; i < 8; ++i) st[t][i] = 0.0f;
      const float* krow = &kbuf[cur][(t * 16 + ln) * LROW + hlf * 16];
      v16h bk0, bk1;
      #pragma unroll
      for (int i = 0; i < 4; ++i) {
        float4 f0 = *(const float4*)(krow + 4 * i);        // d-block 0..31
        float4 f1 = *(const float4*)(krow + 32 + 4 * i);   // d-block 32..63
        bk0[4 * i + 0] = (half_t)f0.x; bk0[4 * i + 1] = (half_t)f0.y;
        bk0[4 * i + 2] = (half_t)f0.z; bk0[4 * i + 3] = (half_t)f0.w;
        bk1[4 * i + 0] = (half_t)f1.x; bk1[4 * i + 1] = (half_t)f1.y;
        bk1[4 * i + 2] = (half_t)f1.z; bk1[4 * i + 3] = (half_t)f1.w;
      }
      st[t] = __builtin_amdgcn_wmma_f32_16x16x32_f16(false, aq0, false, bk0,
                                                     (short)0, st[t], false, false);
      st[t] = __builtin_amdgcn_wmma_f32_16x16x32_f16(false, aq1, false, bk1,
                                                     (short)0, st[t], false, false);
    }

    // ---- mask + diagonal penalty ----
    #pragma unroll
    for (int t = 0; t < 2; ++t) {
      const int keyg = kb0 + t * 16 + ln;
      #pragma unroll
      for (int v = 0; v < 8; ++v) {
        const int qg = qbase + v + 8 * hlf;
        const unsigned char mk = mp[(size_t)qg * SS + keyg];
        float s = mk ? st[t][v] : -__builtin_inff();
        if (qg == keyg) s += -100000.0f;
        st[t][v] = s;
      }
    }

    // ---- online softmax (row reductions within each 16-lane half) ----
    float sc_r[8];
    #pragma unroll
    for (int v = 0; v < 8; ++v) {
      float mx = fmaxf(st[0][v], st[1][v]);
      #pragma unroll
      for (int off = 1; off < 16; off <<= 1) mx = fmaxf(mx, __shfl_xor(mx, off, 32));
      const float mn = fmaxf(m_r[v], mx);
      const float sc = __expf(m_r[v] - mn);
      m_r[v] = mn;
      const float p0 = __expf(st[0][v] - mn);
      const float p1 = __expf(st[1][v] - mn);
      st[0][v] = p0; st[1][v] = p1;
      float rs = p0 + p1;
      #pragma unroll
      for (int off = 1; off < 16; off <<= 1) rs += __shfl_xor(rs, off, 32);
      l_r[v] = l_r[v] * sc + rs;
      sc_r[v] = sc;
    }

    // ---- stage P^T ([q][key] f16) + per-row rescale factor through LDS ----
    #pragma unroll
    for (int v = 0; v < 8; ++v) {
      const int qr = v + 8 * hlf;
      p_buf[wave][qr * 32 + ln]      = (half_t)st[0][v];
      p_buf[wave][qr * 32 + 16 + ln] = (half_t)st[1][v];
      if (ln == v) s_buf[wave][qr] = sc_r[v];
    }
    __builtin_amdgcn_wave_barrier();

    // rescale O^T accumulators: q = ln is per-lane uniform in O^T layout
    const float scq = s_buf[wave][ln];
    #pragma unroll
    for (int cc = 0; cc < 4; ++cc)
      #pragma unroll
      for (int i = 0; i < 8; ++i) oacc[cc][i] *= scq;

    // B operand P^T (K=key within chunk, N=q)
    v16h bp;
    #pragma unroll
    for (int v = 0; v < 8; ++v) {
      h2 pr = *(const h2*)&p_buf[wave][ln * 32 + hlf * 16 + 2 * v];
      bp[2 * v] = pr.x; bp[2 * v + 1] = pr.y;
    }
    __builtin_amdgcn_wave_barrier();

    // ---- O^T += V^T x P^T : Vt gathers from padded LDS, conflict-free ----
    #pragma unroll
    for (int cc = 0; cc < 4; ++cc) {
      v16h av;
      #pragma unroll
      for (int v = 0; v < 8; ++v) {
        const int kl = kpair_base(v, hlf);  // key within chunk
        av[2 * v]     = (half_t)vbuf[cur][kl * LROW + cc * 16 + ln];
        av[2 * v + 1] = (half_t)vbuf[cur][(kl + 1) * LROW + cc * 16 + ln];
      }
      oacc[cc] = __builtin_amdgcn_wmma_f32_16x16x32_f16(false, av, false, bp,
                                                        (short)0, oacc[cc], false, false);
    }

    __syncthreads();  // all waves done reading kbuf/vbuf[cur]
    cur ^= 1;
  }

  // ---- epilogue: broadcast l per q, normalize, coalesced store ----
  #pragma unroll
  for (int v = 0; v < 8; ++v)
    if (ln == v) s_buf[wave][v + 8 * hlf] = l_r[v];
  __builtin_amdgcn_wave_barrier();
  const float inv = 1.0f / s_buf[wave][ln];

  float* orow = O + bh_off + (size_t)(qbase + ln) * RS;
  #pragma unroll
  for (int cc = 0; cc < 4; ++cc) {
    const int d0 = cc * 16 + 8 * hlf;  // 8 consecutive dims per lane
    float4 o0 = {oacc[cc][0] * inv, oacc[cc][1] * inv,
                 oacc[cc][2] * inv, oacc[cc][3] * inv};
    float4 o1 = {oacc[cc][4] * inv, oacc[cc][5] * inv,
                 oacc[cc][6] * inv, oacc[cc][7] * inv};
    *(float4*)(orow + d0)     = o0;
    *(float4*)(orow + d0 + 4) = o1;
  }
}

extern "C" void kernel_launch(void* const* d_in, const int* in_sizes, int n_in,
                              void* d_out, int out_size, void* d_ws, size_t ws_size,
                              hipStream_t stream) {
  (void)in_sizes; (void)n_in; (void)out_size; (void)d_ws; (void)ws_size;
  const float* q = (const float*)d_in[0];
  const float* k = (const float*)d_in[1];
  const float* v = (const float*)d_in[2];
  const unsigned char* m = (const unsigned char*)d_in[3];
  float* out = (float*)d_out;

  const int tiles  = BB * HH * (SS / 16);  // 8192 query tiles
  const int blocks = tiles / WAVES;        // 1024 blocks x 256 threads (8 waves)
  attn_fwd_19164144075426<<<blocks, 256, 0, stream>>>(q, k, v, m, out);
}
